// EdgeGatedGraphConv_31490700214962
// MI455X (gfx1250) — matrix operand order repaired and checked by reading, not compiled, not run on hardware
//
#include <hip/hip_runtime.h>
#include <hip/hip_bf16.h>

// ---------------------------------------------------------------------------
// EdgeGatedGraphConv for MI455X (gfx1250): f16 WMMA (f32 accumulate) GEMMs,
// TDM (tensor_load_to_lds) staging of the big f32 tiles, fused
// gather/gate/scatter edge kernel, wave32 shuffle LayerNorm.
// ---------------------------------------------------------------------------

typedef __attribute__((ext_vector_type(16))) _Float16     v16h;
typedef __attribute__((ext_vector_type(8)))  float        v8f;
typedef __attribute__((ext_vector_type(4)))  unsigned int u32x4;
typedef __attribute__((ext_vector_type(8)))  int          i32x8;
typedef __attribute__((ext_vector_type(4)))  int          i32x4;

#define DIM 128
#define TM  64          // rows per block
#define NCT 8           // 128/16 column tiles

#if defined(__has_builtin)
#if __has_builtin(__builtin_amdgcn_tensor_load_to_lds) && \
    __has_builtin(__builtin_amdgcn_s_wait_tensorcnt)
#define HAVE_TDM 1
#endif
#endif
#ifndef HAVE_TDM
#define HAVE_TDM 0
#endif

// Compile-time diagnostics (visible in hipcc stderr) so the compile-only loop
// can tell which staging path was selected.
#define TDM_STR2(x) #x
#define TDM_STR(x) TDM_STR2(x)
#if HAVE_TDM
#pragma message("CDNA5-PATH: HAVE_TDM=1 (tensor_load_to_lds builtin), clang=" TDM_STR(__clang_major__))
#else
#pragma message("CDNA5-PATH: HAVE_TDM=0 (plain-load fallback), clang=" TDM_STR(__clang_major__))
#endif

#if HAVE_TDM
// Issue one TDM descriptor: DMA a tile_rows x width f32 tile (row stride
// `stride` elements) from global into LDS.  Rows past `rows_avail` read as
// zero (tensor OOB).  D# layout per CDNA5 ISA ch.8.
__device__ __forceinline__ void tdm_load_2d_f32(void* lds_dst, const void* gsrc,
                                                unsigned rows_avail,
                                                unsigned width, unsigned stride,
                                                unsigned tile_rows) {
  unsigned long long ga  = (unsigned long long)gsrc;
  unsigned           lds = (unsigned)(unsigned long long)lds_dst; // LDS byte offset

  u32x4 g0;
  g0[0] = 1u;                                        // count=1, user mode
  g0[1] = lds;                                       // lds_addr
  g0[2] = (unsigned)(ga & 0xffffffffu);              // global_addr[31:0]
  g0[3] = (unsigned)((ga >> 32) & 0x01ffffffu)       // global_addr[56:32]
          | (2u << 30);                              // type=2 ("image")

  i32x8 g1;
  g1[0] = (int)(2u << 16);                           // data_size=2 (4 bytes)
  g1[1] = (int)(width << 16);                        // tensor_dim0[15:0]
  g1[2] = (int)((width >> 16) | (rows_avail << 16)); // dim0 hi | tensor_dim1 lo
  g1[3] = (int)((rows_avail >> 16) | (width << 16)); // dim1 hi | tile_dim0
  g1[4] = (int)tile_rows;                            // tile_dim1 (tile_dim2=0)
  g1[5] = (int)stride;                               // tensor_dim0_stride[31:0]
  g1[6] = 0;                                         // stride hi | dim1_stride lo
  g1[7] = 0;

  i32x4 gz = {0, 0, 0, 0};                           // groups 2/3 unused (2D)
#if __clang_major__ >= 23
  i32x8 gz8 = {0, 0, 0, 0, 0, 0, 0, 0};
  __builtin_amdgcn_tensor_load_to_lds(g0, g1, gz, gz, gz8, 0);
#else
  __builtin_amdgcn_tensor_load_to_lds(g0, g1, gz, gz, 0);
#endif
}
#endif // HAVE_TDM

// ---- WMMA fragment helpers -------------------------------------------------
// A (16x32 f16, MxK), ISA 7.12.2: lanes 0-15 hold M=lane; v0..v3 K=0..7,
// v4..v7 K=16..23; lanes 16-31: K offset +8.
__device__ __forceinline__ v16h load_a_lds(const _Float16* A, int lda, int r0,
                                           int k0, int lane) {
  v16h a;
  const _Float16* row = A + (size_t)(r0 + (lane & 15)) * lda + k0 + ((lane & 16) >> 1);
#pragma unroll
  for (int i = 0; i < 16; ++i)
    a[i] = row[(i & 7) + ((i & 8) << 1)];
  return a;
}

// B fragments pre-packed so each lane's 16 halves are contiguous:
// fragment (kt, ct) lives at ((kt*NCT + ct)*32 + lane)*16.
__device__ __forceinline__ v16h load_b_packed(const _Float16* Wp, int kt, int ct,
                                              int lane) {
  const v16h* p = reinterpret_cast<const v16h*>(Wp);
  return p[(size_t)(kt * NCT + ct) * 32 + lane];
}

// One wave computes a 16-row x 64-col strip: 4 col tiles starting at ct0.
template <int K>
__device__ __forceinline__ void wave_gemm(const _Float16* Ash, int lda, int r0,
                                          const _Float16* Wp, const float* bias,
                                          int ct0, int lane, v8f (&acc)[4]) {
#pragma unroll
  for (int t = 0; t < 4; ++t) {
    float bv = bias[(ct0 + t) * 16 + (lane & 15)];
#pragma unroll
    for (int j = 0; j < 8; ++j) acc[t][j] = bv;
  }
#pragma unroll
  for (int kt = 0; kt < K / 32; ++kt) {
    v16h a = load_a_lds(Ash, lda, r0, kt * 32, lane);
#pragma unroll
    for (int t = 0; t < 4; ++t) {
      v16h b = load_b_packed(Wp, kt, ct0 + t, lane);
      acc[t] = __builtin_amdgcn_wmma_f32_16x16x32_f16(
          false, a, false, b, (short)0, acc[t], false, false);
    }
  }
}

// C/D 16x16 f32: VGPR j -> (M=j, N=lane) lanes 0-15; (M=j+8, N=lane-16).
__device__ __forceinline__ void store_acc_f32(float* O, int ldo, int r0, int ct0,
                                              const v8f (&acc)[4], int lane) {
  int rb = r0 + ((lane & 16) >> 1);
  int n  = lane & 15;
#pragma unroll
  for (int t = 0; t < 4; ++t)
#pragma unroll
    for (int j = 0; j < 8; ++j)
      O[(size_t)(rb + j) * ldo + (ct0 + t) * 16 + n] = acc[t][j];
}

__device__ __forceinline__ void store_acc_silu_f16(_Float16* O, int ldo, int r0,
                                                   int ct0, const v8f (&acc)[4],
                                                   int lane) {
  int rb = r0 + ((lane & 16) >> 1);
  int n  = lane & 15;
#pragma unroll
  for (int t = 0; t < 4; ++t)
#pragma unroll
    for (int j = 0; j < 8; ++j) {
      float x = acc[t][j];
      O[(size_t)(rb + j) * ldo + (ct0 + t) * 16 + n] =
          (_Float16)(x / (1.0f + __expf(-x)));
    }
}

// ---- Weight packing: f32 [K][128] -> f16 fragment-ordered ------------------
__global__ void pack_w_kernel(const float* __restrict__ W, _Float16* __restrict__ dst,
                              int K) {
  int i = blockIdx.x * blockDim.x + threadIdx.x;   // over K*128 halves
  if (i >= K * DIM) return;
  int e    = i & 15;
  int lane = (i >> 4) & 31;
  int tile = i >> 9;               // kt*NCT + ct
  int ct = tile & (NCT - 1);
  int kt = tile >> 3;
  int k = kt * 32 + ((lane & 16) >> 1) + (e & 7) + ((e & 8) << 1);      // lanes>=16: K offset +8 (A-table order)
  int n = ct * 16 + (lane & 15);
  dst[i] = (_Float16)W[(size_t)k * DIM + n];
}

// ---- Node projections: hsp/hdp/msg = h @ W + b -----------------------------
__global__ void node_proj_kernel(const float* __restrict__ h,
                                 const _Float16* __restrict__ Wsrc,
                                 const _Float16* __restrict__ Wdst,
                                 const _Float16* __restrict__ Wmsg,
                                 const float* __restrict__ bsrc,
                                 const float* __restrict__ bdst,
                                 const float* __restrict__ bmsg,
                                 float* __restrict__ hsp, float* __restrict__ hdp,
                                 float* __restrict__ msg, int N) {
  __shared__ _Float16 Ash[TM * DIM];
  int row0 = blockIdx.x * TM;
  for (int i = threadIdx.x; i < TM * DIM; i += blockDim.x) {
    int r = row0 + (i >> 7);
    Ash[i] = (_Float16)(r < N ? h[(size_t)r * DIM + (i & 127)] : 0.0f);
  }
  __syncthreads();
  int wv = threadIdx.x >> 5, lane = threadIdx.x & 31;
  int r0 = (wv & 3) * 16, ct0 = (wv >> 2) * 4;
  const _Float16* Ws[3] = {Wsrc, Wdst, Wmsg};
  const float*    bs[3] = {bsrc, bdst, bmsg};
  float*          os[3] = {hsp, hdp, msg};
#pragma unroll
  for (int m = 0; m < 3; ++m) {
    v8f acc[4];
    wave_gemm<DIM>(Ash, DIM, r0, Ws[m], bs[m], ct0, lane, acc);
    int rb = row0 + r0 + ((lane & 16) >> 1);
    int n  = lane & 15;
#pragma unroll
    for (int t = 0; t < 4; ++t)
#pragma unroll
      for (int j = 0; j < 8; ++j) {
        int r = rb + j;
        if (r < N) os[m][(size_t)r * DIM + (ct0 + t) * 16 + n] = acc[t][j];
      }
  }
}

// ---- Fused edge kernel -----------------------------------------------------
// eg = e@Weg+beg; gate = sigmoid(eg + hsp[src] + hdp[dst]); m = gate*msg[src];
// atomic agg[dst] += m; eu = silu([e|msg_src|msg_dst]@We1+be1)@We2+be2;
// e_new = LN(e + eu).
__global__ void edge_kernel(const float* __restrict__ e,
                            const int* __restrict__ src, const int* __restrict__ dst,
                            const float* __restrict__ hsp, const float* __restrict__ hdp,
                            const float* __restrict__ msg,
                            const _Float16* __restrict__ Weg, const float* __restrict__ beg,
                            const _Float16* __restrict__ We1, const float* __restrict__ be1,
                            const _Float16* __restrict__ We2, const float* __restrict__ be2,
                            const float* __restrict__ g_e, const float* __restrict__ bt_e,
                            float* __restrict__ agg, float* __restrict__ e_out,
                            int E, int N) {
  extern __shared__ char smem[];
  float*    e32 = (float*)smem;                                  // 32 KB
  float*    buf = (float*)(smem + 32768);                        // 32 KB (eg/eu)
  _Float16* ein = (_Float16*)(smem + 65536);                     // 48 KB (K=384)
  _Float16* t16 = (_Float16*)(smem + 65536 + 49152);             // 16 KB
  int*     sidx = (int*)(smem + 65536 + 49152 + 16384);
  int*     didx = sidx + TM;

  int row0 = blockIdx.x * TM;
  int tid  = threadIdx.x;

#if HAVE_TDM
  // Kick off async DMA of the 64x128 f32 e-tile into LDS (overlaps with the
  // msg gathers below).  TDM ignores EXEC; one issue per block.
  if (tid == 0)
    tdm_load_2d_f32(e32, e + (size_t)row0 * DIM, (unsigned)(E - row0),
                    DIM, DIM, TM);
#endif

  for (int i = tid; i < TM; i += blockDim.x) {
    int r = row0 + i;
    int s = (r < E) ? src[r] : 0;
    int d = (r < E) ? dst[r] : 0;
    sidx[i] = (s < 0) ? 0 : (s >= N ? N - 1 : s);
    didx[i] = (d < 0) ? 0 : (d >= N ? N - 1 : d);
  }
  __syncthreads();

  // Gather msg[src] / msg[dst] (L2-resident) into the K=384 f16 A-panel.
  for (int i = tid; i < TM * DIM; i += blockDim.x) {
    int rr = i >> 7, c = i & 127;
    ein[rr * 384 + 128 + c] = (_Float16)msg[(size_t)sidx[rr] * DIM + c];
    ein[rr * 384 + 256 + c] = (_Float16)msg[(size_t)didx[rr] * DIM + c];
  }

#if HAVE_TDM
  __builtin_amdgcn_s_wait_tensorcnt(0);   // no-op for waves that issued nothing
#else
  for (int i = tid; i < TM * DIM; i += blockDim.x) {
    int r = row0 + (i >> 7);
    e32[i] = (r < E) ? e[(size_t)r * DIM + (i & 127)] : 0.0f;
  }
#endif
  __syncthreads();

  // f32 -> f16 conversion of the e tile into the A-panel.
  for (int i = tid; i < TM * DIM; i += blockDim.x) {
    int rr = i >> 7, c = i & 127;
    ein[rr * 384 + c] = (_Float16)e32[i];
  }
  __syncthreads();

  int wv = tid >> 5, lane = tid & 31;
  int r0 = (wv & 3) * 16, ct0 = (wv >> 2) * 4;

  {   // eg GEMM (A = first 128 cols of ein, lda=384)
    v8f acc[4];
    wave_gemm<DIM>(ein, 384, r0, Weg, beg, ct0, lane, acc);
    store_acc_f32(buf, DIM, r0, ct0, acc, lane);
  }
  __syncthreads();

  // gate + scatter-add into agg (L2-resident)
  for (int i = tid; i < TM * DIM; i += blockDim.x) {
    int rr = i >> 7, c = i & 127;
    int r = row0 + rr;
    if (r < E) {
      int s = sidx[rr], d = didx[rr];
      float gl   = buf[i] + hsp[(size_t)s * DIM + c] + hdp[(size_t)d * DIM + c];
      float gate = 1.0f / (1.0f + __expf(-gl));
      float m    = gate * (float)ein[rr * 384 + 128 + c];
      atomicAdd(&agg[(size_t)d * DIM + c], m);
    }
  }
  __syncthreads();

  {   // e1 GEMM over K=384, silu -> t16
    v8f acc[4];
    wave_gemm<384>(ein, 384, r0, We1, be1, ct0, lane, acc);
    store_acc_silu_f16(t16, DIM, r0, ct0, acc, lane);
  }
  __syncthreads();
  {   // e2 GEMM -> buf (eu)
    v8f acc[4];
    wave_gemm<DIM>(t16, DIM, r0, We2, be2, ct0, lane, acc);
    store_acc_f32(buf, DIM, r0, ct0, acc, lane);
  }
  __syncthreads();

  // LayerNorm(e + eu): wave handles 8 rows, lane covers 4 cols each.
  for (int rr = wv * 8; rr < wv * 8 + 8; ++rr) {
    int r = row0 + rr;
    float x[4], s = 0.0f, sq = 0.0f;
#pragma unroll
    for (int j = 0; j < 4; ++j) {
      int c = lane + 32 * j;
      float v = e32[rr * DIM + c] + buf[rr * DIM + c];
      x[j] = v; s += v; sq += v * v;
    }
#pragma unroll
    for (int off = 16; off >= 1; off >>= 1) {
      s  += __shfl_xor(s, off, 32);
      sq += __shfl_xor(sq, off, 32);
    }
    float mu   = s * (1.0f / DIM);
    float var  = sq * (1.0f / DIM) - mu * mu;
    float rstd = rsqrtf(var + 1e-5f);
    if (r < E)
#pragma unroll
      for (int j = 0; j < 4; ++j) {
        int c = lane + 32 * j;
        e_out[(size_t)r * DIM + c] = (x[j] - mu) * rstd * g_e[c] + bt_e[c];
      }
  }
}

// ---- Node update: LN(h + MLP(concat(h, agg))) ------------------------------
__global__ void node_update_kernel(const float* __restrict__ h,
                                   const float* __restrict__ agg,
                                   const _Float16* __restrict__ Wn1, const float* __restrict__ bn1,
                                   const _Float16* __restrict__ Wn2, const float* __restrict__ bn2,
                                   const float* __restrict__ g_n, const float* __restrict__ bt_n,
                                   float* __restrict__ h_out, int N) {
  extern __shared__ char smem[];
  float*    h32 = (float*)smem;                         // 32 KB
  float*    buf = (float*)(smem + 32768);               // 32 KB (agg stage / nu)
  _Float16* ain = (_Float16*)(smem + 65536);            // 32 KB (K=256)
  _Float16* t16 = (_Float16*)(smem + 65536 + 32768);    // 16 KB

  int row0 = blockIdx.x * TM;
  int tid  = threadIdx.x;

#if HAVE_TDM
  if (tid == 0) {
    tdm_load_2d_f32(h32, h   + (size_t)row0 * DIM, (unsigned)(N - row0), DIM, DIM, TM);
    tdm_load_2d_f32(buf, agg + (size_t)row0 * DIM, (unsigned)(N - row0), DIM, DIM, TM);
  }
  __builtin_amdgcn_s_wait_tensorcnt(0);
  __syncthreads();
  for (int i = tid; i < TM * DIM; i += blockDim.x) {
    int rr = i >> 7, c = i & 127;
    ain[rr * 256 + c]       = (_Float16)h32[i];
    ain[rr * 256 + 128 + c] = (_Float16)buf[i];
  }
#else
  for (int i = tid; i < TM * DIM; i += blockDim.x) {
    int rr = i >> 7, c = i & 127;
    int r = row0 + rr;
    float hv = (r < N) ? h[(size_t)r * DIM + c]   : 0.0f;
    float av = (r < N) ? agg[(size_t)r * DIM + c] : 0.0f;
    h32[i] = hv;
    ain[rr * 256 + c]       = (_Float16)hv;
    ain[rr * 256 + 128 + c] = (_Float16)av;
  }
#endif
  __syncthreads();

  int wv = tid >> 5, lane = tid & 31;
  int r0 = (wv & 3) * 16, ct0 = (wv >> 2) * 4;
  {
    v8f acc[4];
    wave_gemm<256>(ain, 256, r0, Wn1, bn1, ct0, lane, acc);
    store_acc_silu_f16(t16, DIM, r0, ct0, acc, lane);
  }
  __syncthreads();
  {
    v8f acc[4];
    wave_gemm<DIM>(t16, DIM, r0, Wn2, bn2, ct0, lane, acc);
    store_acc_f32(buf, DIM, r0, ct0, acc, lane);
  }
  __syncthreads();

  for (int rr = wv * 8; rr < wv * 8 + 8; ++rr) {
    int r = row0 + rr;
    float x[4], s = 0.0f, sq = 0.0f;
#pragma unroll
    for (int j = 0; j < 4; ++j) {
      int c = lane + 32 * j;
      float v = h32[rr * DIM + c] + buf[rr * DIM + c];
      x[j] = v; s += v; sq += v * v;
    }
#pragma unroll
    for (int off = 16; off >= 1; off >>= 1) {
      s  += __shfl_xor(s, off, 32);
      sq += __shfl_xor(sq, off, 32);
    }
    float mu   = s * (1.0f / DIM);
    float var  = sq * (1.0f / DIM) - mu * mu;
    float rstd = rsqrtf(var + 1e-5f);
    if (r < N)
#pragma unroll
      for (int j = 0; j < 4; ++j) {
        int c = lane + 32 * j;
        h_out[(size_t)r * DIM + c] = (x[j] - mu) * rstd * g_n[c] + bt_n[c];
      }
  }
}

// ---------------------------------------------------------------------------
extern "C" void kernel_launch(void* const* d_in, const int* in_sizes, int n_in,
                              void* d_out, int out_size, void* d_ws, size_t ws_size,
                              hipStream_t stream) {
  const float* h     = (const float*)d_in[0];
  const float* e     = (const float*)d_in[1];
  const int*   src   = (const int*)d_in[2];
  const int*   dst   = (const int*)d_in[3];
  const float* w_src = (const float*)d_in[4];  const float* b_src = (const float*)d_in[5];
  const float* w_dst = (const float*)d_in[6];  const float* b_dst = (const float*)d_in[7];
  const float* w_eg  = (const float*)d_in[8];  const float* b_eg  = (const float*)d_in[9];
  const float* w_msg = (const float*)d_in[10]; const float* b_msg = (const float*)d_in[11];
  const float* w_n1  = (const float*)d_in[12]; const float* b_n1  = (const float*)d_in[13];
  const float* w_n2  = (const float*)d_in[14]; const float* b_n2  = (const float*)d_in[15];
  const float* w_e1  = (const float*)d_in[16]; const float* b_e1  = (const float*)d_in[17];
  const float* w_e2  = (const float*)d_in[18]; const float* b_e2  = (const float*)d_in[19];
  const float* g_n   = (const float*)d_in[20]; const float* bt_n  = (const float*)d_in[21];
  const float* g_e   = (const float*)d_in[22]; const float* bt_e  = (const float*)d_in[23];

  int N = in_sizes[0] / DIM;
  int E = in_sizes[1] / DIM;

  // Workspace layout: packed f16 weights (first 512 KB), then f32 node arrays.
  char* ws = (char*)d_ws;
  _Float16* Wsrc = (_Float16*)ws;          // 128x128
  _Float16* Wdst = Wsrc + 16384;
  _Float16* Wmsg = Wdst + 16384;
  _Float16* Weg  = Wmsg + 16384;
  _Float16* Wn1  = Weg  + 16384;           // 256x128
  _Float16* Wn2  = Wn1  + 32768;
  _Float16* We1  = Wn2  + 16384;           // 384x128
  _Float16* We2  = We1  + 49152;
  float* hsp = (float*)(ws + 524288);
  float* hdp = hsp + (size_t)N * DIM;
  float* msg = hdp + (size_t)N * DIM;
  float* agg = msg + (size_t)N * DIM;

  float* h_out = (float*)d_out;
  float* e_out = h_out + (size_t)N * DIM;

  // 1) Pack weights f32 -> f16 fragment order.
  pack_w_kernel<<<(128 * DIM + 255) / 256, 256, 0, stream>>>(w_src, Wsrc, 128);
  pack_w_kernel<<<(128 * DIM + 255) / 256, 256, 0, stream>>>(w_dst, Wdst, 128);
  pack_w_kernel<<<(128 * DIM + 255) / 256, 256, 0, stream>>>(w_msg, Wmsg, 128);
  pack_w_kernel<<<(128 * DIM + 255) / 256, 256, 0, stream>>>(w_eg,  Weg,  128);
  pack_w_kernel<<<(256 * DIM + 255) / 256, 256, 0, stream>>>(w_n1,  Wn1,  256);
  pack_w_kernel<<<(128 * DIM + 255) / 256, 256, 0, stream>>>(w_n2,  Wn2,  128);
  pack_w_kernel<<<(384 * DIM + 255) / 256, 256, 0, stream>>>(w_e1,  We1,  384);
  pack_w_kernel<<<(128 * DIM + 255) / 256, 256, 0, stream>>>(w_e2,  We2,  128);

  // 2) Node projections.
  node_proj_kernel<<<(N + TM - 1) / TM, 256, 0, stream>>>(
      h, Wsrc, Wdst, Wmsg, b_src, b_dst, b_msg, hsp, hdp, msg, N);

  // 3) Zero aggregation buffer (graph-capture-safe async memset).
  hipMemsetAsync(agg, 0, (size_t)N * DIM * sizeof(float), stream);

  // 4) Fused edge kernel.
  size_t edge_smem = 65536 + 49152 + 16384 + 2 * TM * sizeof(int);
  edge_kernel<<<(E + TM - 1) / TM, 256, edge_smem, stream>>>(
      e, src, dst, hsp, hdp, msg, Weg, b_eg, We1, b_e1, We2, b_e2,
      g_e, bt_e, agg, e_out, E, N);

  // 5) Node update.
  size_t node_smem = 65536 + 32768 + 16384;
  node_update_kernel<<<(N + TM - 1) / TM, 256, node_smem, stream>>>(
      h, agg, Wn1, b_n1, Wn2, b_n2, g_n, bt_n, h_out, N);
}
